// MultiHeadedAttention_37409165148608
// MI455X (gfx1250) — matrix operand order repaired and checked by reading, not compiled, not run on hardware
//
#include <hip/hip_runtime.h>
#include <hip/hip_bf16.h>
#include <math.h>

typedef _Float16 half_t;
typedef __attribute__((ext_vector_type(16))) _Float16 v16h;
typedef __attribute__((ext_vector_type(8)))  float    v8f;

// Problem constants
#define NB   8      // batch
#define NS   1024   // sequence
#define ND   512    // model dim
#define NH   8      // heads
#define NDK  64     // d_k == d_v

static __device__ __forceinline__ v8f zero8() {
  v8f z;
#pragma unroll
  for (int i = 0; i < 8; ++i) z[i] = 0.0f;
  return z;
}

static __device__ __forceinline__ v8f wmma16(v16h a, v16h b, v8f c) {
  // D = A(16x32 f16) * B(32x16 f16) + C(16x16 f32)
  return __builtin_amdgcn_wmma_f32_16x16x32_f16(false, a, false, b, (short)0, c,
                                                false, false);
}

// A fragment (16x32, f16). Source: row-major [m][k], leading dim ld (halves).
// Layout: lanes 0-15 hold M=lane, K=0..15; lanes 16-31 hold
// M=lane-16, K=16..31; 2 halves per VGPR in K order.
static __device__ __forceinline__ v16h load_a_frag(const half_t* p, int ld, int lane) {
  const int m  = lane & 15;
  const int kb = (lane >> 4) << 4;  // 0 or 16
  const half_t* row = p + (size_t)m * ld;
  v16h a;
#pragma unroll
  for (int i = 0; i < 16; ++i) a[i] = row[kb + i];
  return a;
}

// B fragment (32x16, f16) where the source holds B^T: rows are N (16 of them),
// cols are K (32), leading dim ld. Lane holds column N=lane&15;
// lanes 0-15: K=0..15, lanes 16-31: K=16..31.  (contiguous -> ds_load_b128)
static __device__ __forceinline__ v16h load_b_frag(const half_t* p, int ld, int lane) {
  const int n  = lane & 15;
  const int kb = (lane >> 4) << 4;  // 0 or 16
  const half_t* row = p + (size_t)n * ld;
  v16h b;
#pragma unroll
  for (int i = 0; i < 16; ++i) b[i] = row[kb + i];
  return b;
}

// ---------------------------------------------------------------------------
// Kernel 1: f32 -> f16 conversion (weights)
// ---------------------------------------------------------------------------
__global__ void convert_f32_f16(const float* __restrict__ src,
                                half_t* __restrict__ dst, int n) {
  int i = blockIdx.x * blockDim.x + threadIdx.x;
  int stride = gridDim.x * blockDim.x;
  for (; i < n; i += stride) dst[i] = (half_t)src[i];
}

// ---------------------------------------------------------------------------
// Kernel 2: C[m][n] = sum_k A[m][k] * W[n][k]
// A: M x 512 f32 (converted to f16 while staging into LDS)
// W: 512 x 512 f16 (row n contiguous over k)
// Block: 256 threads (8 waves), tile 128(M) x 64(N), k-step 32.
// mode 0: out16[((b*8+h)*1024+s)*64+dk]  = (acc+bias)*scale   (f16, row-major)
// mode 2: out16[((b*8+h)*64+dk)*1024+s]  = (acc+bias)*scale   (f16, transposed)
// mode 1: out32[m*512+n]                 = (acc+bias)*scale   (f32)
// ---------------------------------------------------------------------------
__global__ __launch_bounds__(256)
void gemm512(const float* __restrict__ A, const half_t* __restrict__ W,
             const float* __restrict__ bias,
             half_t* __restrict__ out16, float* __restrict__ out32,
             int mode, float scale) {
  __shared__ __align__(16) half_t As[128][32];
  __shared__ __align__(16) half_t Bs[64][32];

  const int lane = threadIdx.x & 31;
  const int wv   = threadIdx.x >> 5;
  const int m0   = blockIdx.x * 128;
  const int n0   = blockIdx.y * 64;

  v8f acc[4];
#pragma unroll
  for (int t = 0; t < 4; ++t) acc[t] = zero8();

  for (int kk = 0; kk < 512; kk += 32) {
    {  // stage A: 128 x 32, two threads per row
      const int r  = threadIdx.x >> 1;
      const int cb = (threadIdx.x & 1) * 16;
      const float* s = A + (size_t)(m0 + r) * 512 + kk + cb;
#pragma unroll
      for (int i = 0; i < 16; ++i) As[r][cb + i] = (half_t)s[i];
    }
    if (threadIdx.x < 128) {  // stage W tile: 64 x 32
      const int r  = threadIdx.x >> 1;
      const int cb = (threadIdx.x & 1) * 16;
      const half_t* s = W + (size_t)(n0 + r) * 512 + kk + cb;
#pragma unroll
      for (int i = 0; i < 16; ++i) Bs[r][cb + i] = s[i];
    }
    __syncthreads();

    v16h a = load_a_frag(&As[wv * 16][0], 32, lane);
#pragma unroll
    for (int t = 0; t < 4; ++t) {
      v16h b = load_b_frag(&Bs[t * 16][0], 32, lane);
      acc[t] = wmma16(a, b, acc[t]);
    }
    __syncthreads();
  }

  const int hi  = lane >> 4;
  const int col = lane & 15;
#pragma unroll
  for (int t = 0; t < 4; ++t) {
    const int n  = n0 + t * 16 + col;
    const float bvv = bias[n];
#pragma unroll
    for (int j = 0; j < 8; ++j) {
      const int m = m0 + wv * 16 + j + 8 * hi;
      const float v = (acc[t][j] + bvv) * scale;
      if (mode == 0) {
        const int b = m >> 10, s = m & 1023;
        const int h = n >> 6, dk = n & 63;
        out16[(((size_t)(b * NH + h)) * NS + s) * NDK + dk] = (half_t)v;
      } else if (mode == 2) {
        const int b = m >> 10, s = m & 1023;
        const int h = n >> 6, dk = n & 63;
        // transposed: consecutive j -> consecutive s (vectorizable stores)
        out16[(((size_t)(b * NH + h)) * NDK + dk) * NS + s] = (half_t)v;
      } else {
        out32[(size_t)m * 512 + n] = v;
      }
    }
  }
}

// ---------------------------------------------------------------------------
// Kernel 3: fused attention (flash style, online softmax in f32)
// grid: (S/64, 4 head-pairs, B); block: 256 (8 waves)
// wave -> (head = 2*hp + (wv>>2), 16-row q subtile = wv&3)
// Head pairs share one code map tile read; V is pre-transposed so all LDS
// fragment loads are contiguous b128.
// ---------------------------------------------------------------------------
__global__ __launch_bounds__(256)
void attn_kernel(const half_t* __restrict__ q16, const half_t* __restrict__ k16,
                 const half_t* __restrict__ vT16,
                 const float* __restrict__ map_it,
                 const float* __restrict__ map_ist,
                 const float* __restrict__ map_df,
                 const unsigned char* __restrict__ mask,
                 float* __restrict__ ctx) {
  __shared__ __align__(16) half_t kt[2][64][64];   // [head][s][dk]
  __shared__ __align__(16) half_t vt[2][64][64];   // [head][dv][s]  (transposed)
  __shared__ __align__(16) half_t pt[8][16][64];   // per-wave P tile

  const int lane = threadIdx.x & 31;
  const int wv   = threadIdx.x >> 5;
  const int hl   = wv >> 2;        // head within the pair
  const int qsub = wv & 3;
  const int qblk = blockIdx.x;     // 0..15
  const int hp   = blockIdx.y;     // 0..3  (head-group / pair)
  const int b    = blockIdx.z;     // 0..7
  const int head = hp * 2 + hl;
  const int q0   = qblk * 64 + qsub * 16;
  const int hi   = lane >> 4;
  const int col  = lane & 15;

  // Resident Q fragments (16 x 64 per wave; q already scaled by 1/sqrt(dk))
  const half_t* qbase = q16 + (((size_t)(b * NH + head)) * NS + q0) * NDK;
  const v16h qa0 = load_a_frag(qbase, NDK, lane);
  const v16h qa1 = load_a_frag(qbase + 32, NDK, lane);

  const float* map = (hp == 0) ? map_it : (hp == 1) ? map_ist : map_df;

  v8f o[4];
  float mrow[8], lrow[8];
#pragma unroll
  for (int t = 0; t < 4; ++t) o[t] = zero8();
#pragma unroll
  for (int j = 0; j < 8; ++j) { mrow[j] = -3.0e38f; lrow[j] = 0.0f; }

  // staging assignment: 256 threads cover 2 heads x 64 rows x 2 half-rows
  const int sh   = threadIdx.x >> 7;
  const int srow = (threadIdx.x >> 1) & 63;
  const int scol = (threadIdx.x & 1) * 32;

  for (int kb = 0; kb < 16; ++kb) {
    const int ks0 = kb * 64;
    {  // stage K rows and transposed-V rows, 64B vector copies
      const size_t kro =
          (((size_t)(b * NH + hp * 2 + sh)) * NS + ks0 + srow) * NDK + scol;
      const size_t vro =
          (((size_t)(b * NH + hp * 2 + sh)) * NDK + srow) * NS + ks0 + scol;
      const uint4* ks4 = (const uint4*)(k16 + kro);
      const uint4* vs4 = (const uint4*)(vT16 + vro);
      uint4* kd4 = (uint4*)&kt[sh][srow][scol];
      uint4* vd4 = (uint4*)&vt[sh][srow][scol];
#pragma unroll
      for (int i = 0; i < 4; ++i) { kd4[i] = ks4[i]; vd4[i] = vs4[i]; }
    }
    // prefetch next k-block's code-map tile (global_prefetch_b8)
    if (hp < 3 && kb < 15) {
      const int pr = threadIdx.x >> 2;          // 0..63 q-rows of this block
      const int pc = (threadIdx.x & 3) * 16;    // 4 x 64B segments per row
      __builtin_prefetch(
          map + ((size_t)b * NS + qblk * 64 + pr) * NS + (ks0 + 64) + pc, 0, 1);
    }
    __syncthreads();

    // S = q @ k^T  (4 n-subtiles x 2 k-steps)
    v8f sf[4];
#pragma unroll
    for (int t = 0; t < 4; ++t) {
      v8f c = zero8();
      c = wmma16(qa0, load_b_frag(&kt[hl][t * 16][0], 64, lane), c);
      c = wmma16(qa1, load_b_frag(&kt[hl][t * 16][32], 64, lane), c);
      sf[t] = c;
    }

    // Per-head-group score modification + mask
#pragma unroll
    for (int t = 0; t < 4; ++t) {
      const int kc = ks0 + t * 16 + col;
      const bool msk = mask[b * NS + kc] != 0;
      if (hp < 2) {
#pragma unroll
        for (int j = 0; j < 8; ++j) {
          const int qr = q0 + j + 8 * hi;
          const float mv = map[((size_t)b * NS + qr) * NS + kc];
          sf[t][j] += -1.0e9f * mv;
        }
      } else if (hp == 2) {
#pragma unroll
        for (int j = 0; j < 8; ++j) {
          const int qr = q0 + j + 8 * hi;
          const float mv = map[((size_t)b * NS + qr) * NS + kc];
          sf[t][j] *= (1.0f + 5.0f * mv);
        }
      }
      if (msk) {
#pragma unroll
        for (int j = 0; j < 8; ++j) sf[t][j] = -1.0e18f;
      }
    }

    // Online softmax update per row j (row spans 16 lanes x 4 subtiles)
#pragma unroll
    for (int j = 0; j < 8; ++j) {
      float mx = sf[0][j];
#pragma unroll
      for (int t = 1; t < 4; ++t) mx = fmaxf(mx, sf[t][j]);
#pragma unroll
      for (int d = 1; d < 16; d <<= 1) mx = fmaxf(mx, __shfl_xor(mx, d, 32));
      const float mnew = fmaxf(mrow[j], mx);
      const float corr = __expf(mrow[j] - mnew);
      float psum = 0.0f;
#pragma unroll
      for (int t = 0; t < 4; ++t) {
        const float p = __expf(sf[t][j] - mnew);
        sf[t][j] = p;
        psum += p;
      }
#pragma unroll
      for (int d = 1; d < 16; d <<= 1) psum += __shfl_xor(psum, d, 32);
      lrow[j] = lrow[j] * corr + psum;
      mrow[j] = mnew;
#pragma unroll
      for (int t = 0; t < 4; ++t) o[t][j] *= corr;
    }

    // P (C-layout) -> per-wave LDS tile (row-major 16x64, f16)
#pragma unroll
    for (int t = 0; t < 4; ++t) {
#pragma unroll
      for (int j = 0; j < 8; ++j)
        pt[wv][j + 8 * hi][t * 16 + col] = (half_t)sf[t][j];
    }

    // O += P @ V : B fragments come from the transposed V tile, so lane n
    // (= dv) reads contiguous halves over k (= s) -> ds_load_b128.
    v16h pa0 = load_a_frag(&pt[wv][0][0], 64, lane);
    v16h pa1 = load_a_frag(&pt[wv][0][32], 64, lane);
#pragma unroll
    for (int t = 0; t < 4; ++t) {
      o[t] = wmma16(pa0, load_b_frag(&vt[hl][t * 16][0], 64, lane), o[t]);
      o[t] = wmma16(pa1, load_b_frag(&vt[hl][t * 16][32], 64, lane), o[t]);
    }
    __syncthreads();
  }

  // Normalize and store ctx as (B, S, H*DV) f32 for the output projection
#pragma unroll
  for (int t = 0; t < 4; ++t) {
#pragma unroll
    for (int j = 0; j < 8; ++j) {
      const int qr = q0 + j + 8 * hi;
      ctx[((size_t)b * NS + qr) * ND + head * NDK + t * 16 + col] =
          o[t][j] / lrow[j];
    }
  }
}

// ---------------------------------------------------------------------------
extern "C" void kernel_launch(void* const* d_in, const int* in_sizes, int n_in,
                              void* d_out, int out_size, void* d_ws, size_t ws_size,
                              hipStream_t stream) {
  (void)in_sizes; (void)n_in; (void)out_size; (void)ws_size;
  const float* key   = (const float*)d_in[0];
  const float* value = (const float*)d_in[1];
  const float* query = (const float*)d_in[2];
  const float* m_it  = (const float*)d_in[3];
  const float* m_ist = (const float*)d_in[4];
  const float* m_df  = (const float*)d_in[5];
  const unsigned char* mask = (const unsigned char*)d_in[6];
  const float* Wq = (const float*)d_in[7];
  const float* bq = (const float*)d_in[8];
  const float* Wk = (const float*)d_in[9];
  const float* bk = (const float*)d_in[10];
  const float* Wv = (const float*)d_in[11];
  const float* bv = (const float*)d_in[12];
  const float* Wo = (const float*)d_in[13];
  const float* bo = (const float*)d_in[14];
  float* out = (float*)d_out;

  // Workspace layout (halves): 4 weight mats (256Kh each), q/k/vT (4Mh each), ctx f32
  half_t* wsH  = (half_t*)d_ws;
  half_t* wq16 = wsH;
  half_t* wk16 = wq16 + 262144;
  half_t* wv16 = wk16 + 262144;
  half_t* wo16 = wv16 + 262144;
  half_t* q16  = wo16 + 262144;
  half_t* k16  = q16 + 4194304;
  half_t* vT16 = k16 + 4194304;
  float*  ctx  = (float*)(vT16 + 4194304);

  convert_f32_f16<<<256, 256, 0, stream>>>(Wq, wq16, 262144);
  convert_f32_f16<<<256, 256, 0, stream>>>(Wk, wk16, 262144);
  convert_f32_f16<<<256, 256, 0, stream>>>(Wv, wv16, 262144);
  convert_f32_f16<<<256, 256, 0, stream>>>(Wo, wo16, 262144);

  dim3 g(64, 8);  // (8192/128) x (512/64)
  gemm512<<<g, 256, 0, stream>>>(query, wq16, bq, q16, nullptr, 0, 0.125f);
  gemm512<<<g, 256, 0, stream>>>(key,   wk16, bk, k16, nullptr, 0, 1.0f);
  gemm512<<<g, 256, 0, stream>>>(value, wv16, bv, vT16, nullptr, 2, 1.0f);

  attn_kernel<<<dim3(16, 4, 8), 256, 0, stream>>>(q16, k16, vT16, m_it, m_ist,
                                                  m_df, mask, ctx);

  gemm512<<<g, 256, 0, stream>>>(ctx, wo16, bo, nullptr, out, 1, 1.0f);
}
